// CrossAttentionBlock_34445637714693
// MI455X (gfx1250) — compile-verified, matrix operand order checked by reading, not run on hardware
//
#include <hip/hip_runtime.h>
#include <hip/hip_bf16.h>

typedef _Float16 h8   __attribute__((ext_vector_type(8)));
typedef _Float16 v16h __attribute__((ext_vector_type(16)));
typedef float    v8f  __attribute__((ext_vector_type(8)));
typedef unsigned int u32x4 __attribute__((ext_vector_type(4)));
typedef int          i32x4 __attribute__((ext_vector_type(4)));
typedef int          i32x8 __attribute__((ext_vector_type(8)));

#define SEQ   2048
#define CCH   512
#define NH    8
#define DH    64

#if defined(__has_builtin)
#if __has_builtin(__builtin_amdgcn_tensor_load_to_lds)
#define USE_TDM 1
#endif
#endif
#ifndef USE_TDM
#define USE_TDM 0
#endif

// ---------------------------------------------------------------------------
// Fragment loader for V_WMMA_F32_16X16X32_F16 operands (ISA 7.12.2, 16-bit A):
//   lanes 0-15 : row M = lane,    K runs {k0..k0+7} and {k0+16..k0+23}
//   lanes16-31 : row M = lane-16, K runs {k0+8..+15} and {k0+24..+31}
// Works for global and LDS base pointers (addrspace inferred after inlining).
// ---------------------------------------------------------------------------
__device__ __forceinline__ v16h load_frag16(const _Float16* base, int ld) {
    const int lane = threadIdx.x & 31;
    const int r    = lane & 15;
    const int hi   = lane >> 4;
    const _Float16* p = base + (size_t)r * ld + (hi << 3);
    h8 lo = *(const h8*)(p);
    h8 hp = *(const h8*)(p + 16);
    return __builtin_shufflevector(lo, hp, 0,1,2,3,4,5,6,7,8,9,10,11,12,13,14,15);
}

__device__ __forceinline__ v8f wmma_f16(v16h a, v16h b, v8f c) {
    return __builtin_amdgcn_wmma_f32_16x16x32_f16(
        false, a, false, b, (short)0, c, false, false);
}

#if USE_TDM
// ---------------------------------------------------------------------------
// Tensor Data Mover: 2D f16 tile (tile_d1 rows x tile_d0 elems) global -> LDS.
// D# per ISA ch8: group0 = {count=1, lds_addr, global_addr[56:0], type=2},
// group1 = {wg_mask=0, data_size=1(2B), tensor_dim0/1, tile_dim0/1, stride0}.
// Issue from ONE wave only (EXEC ignored); tracked by TENSORcnt.
// ---------------------------------------------------------------------------
__device__ __forceinline__ void tdm_load_2d(const void* gptr, void* lptr,
                                            unsigned tensor_d0, unsigned tensor_d1,
                                            unsigned tile_d0, unsigned tile_d1,
                                            unsigned stride0) {
    const unsigned long long ga = (unsigned long long)(uintptr_t)gptr;
    const unsigned lds = (unsigned)(uintptr_t)lptr;     // addr[31:0] = LDS offset
    u32x4 g0;
    g0[0] = 1u;                                         // count=1, user mode
    g0[1] = lds;                                        // lds_addr (bytes)
    g0[2] = (unsigned)(ga & 0xffffffffu);               // global_addr[31:0]
    g0[3] = (unsigned)((ga >> 32) & 0x01ffffffu) | (2u << 30);  // addr[56:32] | type=2
    i32x8 g1;
    g1[0] = (1 << 16);                                  // wg_mask=0, data_size=1 (2B)
    g1[1] = (int)((tensor_d0 & 0xffffu) << 16);         // tensor_dim0[15:0] @ bit48
    g1[2] = (int)(((tensor_d0 >> 16) & 0xffffu) | ((tensor_d1 & 0xffffu) << 16));
    g1[3] = (int)(((tensor_d1 >> 16) & 0xffffu) | ((tile_d0 & 0xffffu) << 16));
    g1[4] = (int)(tile_d1 & 0xffffu);                   // tile_dim1 | tile_dim2=0
    g1[5] = (int)stride0;                               // tensor_dim0_stride[31:0]
    g1[6] = 0;                                          // stride0 hi | stride1 lo
    g1[7] = 0;
    i32x4 z4 = {};
#if __clang_major__ >= 23
    i32x8 z8 = {};
    __builtin_amdgcn_tensor_load_to_lds(g0, g1, z4, z4, z8, 0);
#else
    __builtin_amdgcn_tensor_load_to_lds(g0, g1, z4, z4, 0);
#endif
}
#endif

// ---------------------------------------------------------------------------
// Weight transpose + f32 -> f16:  WT[o][i] = W[i][o]
// ---------------------------------------------------------------------------
__global__ void wt_kernel(const float* __restrict__ W, _Float16* __restrict__ WT) {
    int idx = blockIdx.x * 256 + threadIdx.x;
    int o = idx >> 9, i = idx & 511;
    WT[(size_t)o * CCH + i] = (_Float16)W[(size_t)i * CCH + o];
}

// ---------------------------------------------------------------------------
// LayerNorm over channels; x is [b][c][t]. One wave = 32 tokens, coalesced.
// ---------------------------------------------------------------------------
__global__ void ln_kernel(const float* __restrict__ x,
                          const float* __restrict__ gamma,
                          const float* __restrict__ beta,
                          _Float16* __restrict__ XN) {
    const int lane = threadIdx.x & 31;
    const int wave = threadIdx.x >> 5;
    const int t0   = (blockIdx.x * 4 + wave) * 32;
    const int b    = t0 >> 11;
    const int tt   = (t0 & 2047) + lane;
    const float* xb = x + (size_t)b * CCH * SEQ;

    float sum = 0.f, ss = 0.f;
    for (int ch = 0; ch < CCH; ++ch) {
        float v = xb[(size_t)ch * SEQ + tt];
        sum += v; ss += v * v;
    }
    const float mu   = sum * (1.f / CCH);
    const float var  = ss * (1.f / CCH) - mu * mu;
    const float rinv = rsqrtf(var + 1e-5f);

    _Float16* dst = XN + (size_t)(t0 + lane) * CCH;
    for (int ch = 0; ch < CCH; ++ch) {
        float v  = xb[(size_t)ch * SEQ + tt];
        float xn = (v - mu) * rinv * gamma[ch] + beta[ch];
        dst[ch] = (_Float16)xn;
    }
}

// ---------------------------------------------------------------------------
// QKV projection, software-pipelined k-loop (prefetch k+32 fragments into
// registers while current WMMAs execute -> partial s_wait_loadcnt).
// ---------------------------------------------------------------------------
__global__ void qkv_kernel(const _Float16* __restrict__ XN,
                           const _Float16* __restrict__ WTall,
                           _Float16* __restrict__ Q,
                           _Float16* __restrict__ K,
                           _Float16* __restrict__ Vt) {
    const int wave = threadIdx.x >> 5;
    const int lane = threadIdx.x & 31;
    const int r = lane & 15, hi = lane >> 4;
    const int m0   = blockIdx.x * 64 + wave * 16;
    const int col0 = blockIdx.y * 64;
    const int mat  = blockIdx.z;                        // 0=Q 1=K 2=V
    const _Float16* WT = WTall + (size_t)mat * CCH * CCH;

    const _Float16* Ar = XN + (size_t)m0 * CCH;
    const _Float16* B0 = WT + (size_t)(col0 +  0) * CCH;
    const _Float16* B1 = WT + (size_t)(col0 + 16) * CCH;
    const _Float16* B2 = WT + (size_t)(col0 + 32) * CCH;
    const _Float16* B3 = WT + (size_t)(col0 + 48) * CCH;

    v8f acc0 = {}, acc1 = {}, acc2 = {}, acc3 = {};
    v16h a  = load_frag16(Ar, CCH);
    v16h b0 = load_frag16(B0, CCH), b1 = load_frag16(B1, CCH);
    v16h b2 = load_frag16(B2, CCH), b3 = load_frag16(B3, CCH);
#pragma unroll 4
    for (int k0 = 0; k0 < CCH; k0 += 32) {
        const int k1 = (k0 + 32) & (CCH - 1);           // wraps on last iter (unused)
        v16h an  = load_frag16(Ar + k1, CCH);
        v16h bn0 = load_frag16(B0 + k1, CCH);
        v16h bn1 = load_frag16(B1 + k1, CCH);
        v16h bn2 = load_frag16(B2 + k1, CCH);
        v16h bn3 = load_frag16(B3 + k1, CCH);
        acc0 = wmma_f16(a, b0, acc0);
        acc1 = wmma_f16(a, b1, acc1);
        acc2 = wmma_f16(a, b2, acc2);
        acc3 = wmma_f16(a, b3, acc3);
        a = an; b0 = bn0; b1 = bn1; b2 = bn2; b3 = bn3;
    }
    const v8f accs[4] = {acc0, acc1, acc2, acc3};
#pragma unroll
    for (int j = 0; j < 4; ++j) {
        const int cc = col0 + 16 * j + r;
        const int h = cc >> 6, dh = cc & 63;
#pragma unroll
        for (int i = 0; i < 8; ++i) {
            const int g = m0 + i + 8 * hi;
            const int b = g >> 11, t = g & 2047;
            const float v = accs[j][i];
            if (mat == 0)
                Q[((size_t)(b * NH + h) * SEQ + t) * DH + dh] = (_Float16)(v * 0.125f);
            else if (mat == 1)
                K[((size_t)(b * NH + h) * SEQ + t) * DH + dh] = (_Float16)v;
            else
                Vt[((size_t)(b * NH + h) * DH + dh) * SEQ + t] = (_Float16)v;
        }
    }
}

// ---------------------------------------------------------------------------
// Flash attention. 64-query tile per block (4 waves x 16 rows).
// K/V 64-key chunks are staged into double-buffered LDS by the Tensor Data
// Mover (wave 0 issues tensor_load_to_lds; s_wait_tensorcnt(2) overlaps the
// DMA of chunk i+1 with WMMA on chunk i). All waves read WMMA fragments from
// LDS, online softmax in per-wave LDS, fp32 O accumulators.
// ---------------------------------------------------------------------------
__global__ void attn_kernel(const _Float16* __restrict__ Q,
                            const _Float16* __restrict__ K,
                            const _Float16* __restrict__ Vt,
                            _Float16* __restrict__ AO) {
    __shared__ __align__(16) _Float16 sK[2][64][64];   // key-major  [key][dh]
    __shared__ __align__(16) _Float16 sV[2][64][64];   // dh-major   [dh][key]
    __shared__ __align__(16) float    sS[4][16][64];
    __shared__ __align__(16) _Float16 sP[4][16][64];
    __shared__ float sM[4][16], sL[4][16], sA[4][16];

    const int wave = threadIdx.x >> 5;
    const int lane = threadIdx.x & 31;
    const int r = lane & 15, hi = lane >> 4;
    const int q0 = blockIdx.x * 64 + wave * 16;
    const int h  = blockIdx.y;
    const int b  = blockIdx.z;
    const size_t headQK = (size_t)(b * NH + h) * SEQ * DH;
    const size_t headV  = (size_t)(b * NH + h) * DH * SEQ;
    const _Float16* Kh = K  + headQK;
    const _Float16* Vh = Vt + headV;

    const v16h aQ0 = load_frag16(Q + headQK + (size_t)q0 * DH + 0,  DH);
    const v16h aQ1 = load_frag16(Q + headQK + (size_t)q0 * DH + 32, DH);

    v8f o0 = {}, o1 = {}, o2 = {}, o3 = {};
    if (lane < 16) { sM[wave][lane] = -1e30f; sL[wave][lane] = 0.f; }

#if USE_TDM
    if (wave == 0) {   // prime buffer 0
        tdm_load_2d(Kh, &sK[0][0][0], DH, SEQ, DH, 64, DH);
        tdm_load_2d(Vh, &sV[0][0][0], SEQ, DH, 64, DH, SEQ);
    }
#endif
    __syncthreads();

    for (int kv = 0; kv < SEQ; kv += 64) {
#if USE_TDM
        const int buf = (kv >> 6) & 1;
        if (wave == 0) {
            if (kv + 64 < SEQ) {   // issue next chunk, wait only for current
                tdm_load_2d(Kh + (size_t)(kv + 64) * DH, &sK[buf ^ 1][0][0],
                            DH, SEQ, DH, 64, DH);
                tdm_load_2d(Vh + (kv + 64), &sV[buf ^ 1][0][0],
                            SEQ, DH, 64, DH, SEQ);
                __builtin_amdgcn_s_wait_tensorcnt(2);
            } else {
                __builtin_amdgcn_s_wait_tensorcnt(0);
            }
        }
        __syncthreads();
#else
        const int buf = 0;
        for (int it = threadIdx.x; it < 64 * 64 / 8; it += 128) {
            const int row = it >> 3, c8 = (it & 7) << 3;
            *(h8*)(&sK[0][row][c8]) = *(const h8*)(Kh + (size_t)(kv + row) * DH + c8);
            *(h8*)(&sV[0][row][c8]) = *(const h8*)(Vh + (size_t)row * SEQ + kv + c8);
        }
        __syncthreads();
#endif
        // ----- S = Q K^T (scale folded into Q), fragments from LDS -----
        v8f s0 = {}, s1 = {}, s2 = {}, s3 = {};
        s0 = wmma_f16(aQ0, load_frag16(&sK[buf][ 0][0],  64), s0);
        s0 = wmma_f16(aQ1, load_frag16(&sK[buf][ 0][32], 64), s0);
        s1 = wmma_f16(aQ0, load_frag16(&sK[buf][16][0],  64), s1);
        s1 = wmma_f16(aQ1, load_frag16(&sK[buf][16][32], 64), s1);
        s2 = wmma_f16(aQ0, load_frag16(&sK[buf][32][0],  64), s2);
        s2 = wmma_f16(aQ1, load_frag16(&sK[buf][32][32], 64), s2);
        s3 = wmma_f16(aQ0, load_frag16(&sK[buf][48][0],  64), s3);
        s3 = wmma_f16(aQ1, load_frag16(&sK[buf][48][32], 64), s3);

        // spill S fragments (C layout: row = i + 8*hi, col = 16*j + r)
#pragma unroll
        for (int i = 0; i < 8; ++i) {
            const int row = i + 8 * hi;
            sS[wave][row][ 0 + r] = s0[i];
            sS[wave][row][16 + r] = s1[i];
            sS[wave][row][32 + r] = s2[i];
            sS[wave][row][48 + r] = s3[i];
        }
        __syncthreads();

        // ----- online softmax: 2 lanes per row, 32 cols each -----
        const int row = lane >> 1, halfsel = lane & 1;
        const float* srow = &sS[wave][row][halfsel * 32];
        float mx = -1e30f;
        for (int c2 = 0; c2 < 32; ++c2) mx = fmaxf(mx, srow[c2]);
        mx = fmaxf(mx, __shfl_xor(mx, 1));
        const float mold = sM[wave][row];
        const float mnew = fmaxf(mold, mx);
        const float alpha = __expf(mold - mnew);
        float sum = 0.f;
        _Float16* prow = &sP[wave][row][halfsel * 32];
        for (int c2 = 0; c2 < 32; ++c2) {
            float e = __expf(srow[c2] - mnew);
            prow[c2] = (_Float16)e;
            sum += e;
        }
        sum += __shfl_xor(sum, 1);
        if (halfsel == 0) {
            sM[wave][row] = mnew;
            sL[wave][row] = sL[wave][row] * alpha + sum;
            sA[wave][row] = alpha;
        }
        __syncthreads();

        // ----- rescale O, then O += P @ V (fragments from LDS) -----
#pragma unroll
        for (int i = 0; i < 8; ++i) {
            const float av = sA[wave][i + 8 * hi];
            o0[i] *= av; o1[i] *= av; o2[i] *= av; o3[i] *= av;
        }
        const v16h aP0 = load_frag16(&sP[wave][0][0],  64);
        const v16h aP1 = load_frag16(&sP[wave][0][32], 64);
        o0 = wmma_f16(aP0, load_frag16(&sV[buf][ 0][0],  64), o0);
        o0 = wmma_f16(aP1, load_frag16(&sV[buf][ 0][32], 64), o0);
        o1 = wmma_f16(aP0, load_frag16(&sV[buf][16][0],  64), o1);
        o1 = wmma_f16(aP1, load_frag16(&sV[buf][16][32], 64), o1);
        o2 = wmma_f16(aP0, load_frag16(&sV[buf][32][0],  64), o2);
        o2 = wmma_f16(aP1, load_frag16(&sV[buf][32][32], 64), o2);
        o3 = wmma_f16(aP0, load_frag16(&sV[buf][48][0],  64), o3);
        o3 = wmma_f16(aP1, load_frag16(&sV[buf][48][32], 64), o3);
        __syncthreads();
    }

    // ----- epilogue: normalize by l and emit AO[b*seq][c] f16 -----
#pragma unroll
    for (int i = 0; i < 8; ++i) {
        const float linv = 1.f / sL[wave][i + 8 * hi];
        const int g = (b << 11) + q0 + i + 8 * hi;
        _Float16* dst = AO + (size_t)g * CCH + h * DH;
        dst[ 0 + r] = (_Float16)(o0[i] * linv);
        dst[16 + r] = (_Float16)(o1[i] * linv);
        dst[32 + r] = (_Float16)(o2[i] * linv);
        dst[48 + r] = (_Float16)(o3[i] * linv);
    }
}

// ---------------------------------------------------------------------------
// Output projection + bias + residual + transpose back to [b][c][d][h][w].
// Same software-pipelined k-loop as qkv_kernel.
// ---------------------------------------------------------------------------
__global__ void oproj_kernel(const _Float16* __restrict__ AO,
                             const _Float16* __restrict__ WoT,
                             const float* __restrict__ bo,
                             const float* __restrict__ x,
                             float* __restrict__ out) {
    const int wave = threadIdx.x >> 5;
    const int lane = threadIdx.x & 31;
    const int r = lane & 15, hi = lane >> 4;
    const int m0   = blockIdx.x * 64 + wave * 16;
    const int col0 = blockIdx.y * 64;

    const _Float16* Ar = AO + (size_t)m0 * CCH;
    const _Float16* B0 = WoT + (size_t)(col0 +  0) * CCH;
    const _Float16* B1 = WoT + (size_t)(col0 + 16) * CCH;
    const _Float16* B2 = WoT + (size_t)(col0 + 32) * CCH;
    const _Float16* B3 = WoT + (size_t)(col0 + 48) * CCH;

    v8f acc0 = {}, acc1 = {}, acc2 = {}, acc3 = {};
    v16h a  = load_frag16(Ar, CCH);
    v16h b0 = load_frag16(B0, CCH), b1 = load_frag16(B1, CCH);
    v16h b2 = load_frag16(B2, CCH), b3 = load_frag16(B3, CCH);
#pragma unroll 4
    for (int k0 = 0; k0 < CCH; k0 += 32) {
        const int k1 = (k0 + 32) & (CCH - 1);
        v16h an  = load_frag16(Ar + k1, CCH);
        v16h bn0 = load_frag16(B0 + k1, CCH);
        v16h bn1 = load_frag16(B1 + k1, CCH);
        v16h bn2 = load_frag16(B2 + k1, CCH);
        v16h bn3 = load_frag16(B3 + k1, CCH);
        acc0 = wmma_f16(a, b0, acc0);
        acc1 = wmma_f16(a, b1, acc1);
        acc2 = wmma_f16(a, b2, acc2);
        acc3 = wmma_f16(a, b3, acc3);
        a = an; b0 = bn0; b1 = bn1; b2 = bn2; b3 = bn3;
    }
    const v8f accs[4] = {acc0, acc1, acc2, acc3};
#pragma unroll
    for (int j = 0; j < 4; ++j) {
        const int cc = col0 + 16 * j + r;
        const float bias = bo[cc];
#pragma unroll
        for (int i = 0; i < 8; ++i) {
            const int g = m0 + i + 8 * hi;
            const int b = g >> 11, t = g & 2047;
            const size_t idx = (size_t)(b * CCH + cc) * SEQ + t;
            out[idx] = accs[j][i] + bias + x[idx];
        }
    }
}

// ---------------------------------------------------------------------------
// kernel_launch: inputs = {x, Wq, Wk, Wv, Wo, bo, gamma, beta}, all fp32.
// Workspace (f16): XN 4MiB | WT(q,k,v,o) 2MiB | Q 4MiB | K 4MiB | Vt 4MiB |
//                  AO 4MiB  -> 22 MiB total (fits easily in 192 MB L2).
// ---------------------------------------------------------------------------
extern "C" void kernel_launch(void* const* d_in, const int* in_sizes, int n_in,
                              void* d_out, int out_size, void* d_ws, size_t ws_size,
                              hipStream_t stream) {
    const float* x     = (const float*)d_in[0];
    const float* Wq    = (const float*)d_in[1];
    const float* Wk    = (const float*)d_in[2];
    const float* Wv    = (const float*)d_in[3];
    const float* Wo    = (const float*)d_in[4];
    const float* bo    = (const float*)d_in[5];
    const float* gamma = (const float*)d_in[6];
    const float* beta  = (const float*)d_in[7];
    float* out = (float*)d_out;

    char* ws = (char*)d_ws;
    const size_t MiB = 1u << 20;
    _Float16* XN = (_Float16*)(ws + 0 * MiB);
    _Float16* WT = (_Float16*)(ws + 4 * MiB);
    _Float16* Qb = (_Float16*)(ws + 6 * MiB);
    _Float16* Kb = (_Float16*)(ws + 10 * MiB);
    _Float16* Vt = (_Float16*)(ws + 14 * MiB);
    _Float16* AO = (_Float16*)(ws + 18 * MiB);
    const size_t WSZ = (size_t)CCH * CCH;

    wt_kernel<<<1024, 256, 0, stream>>>(Wq, WT + 0 * WSZ);
    wt_kernel<<<1024, 256, 0, stream>>>(Wk, WT + 1 * WSZ);
    wt_kernel<<<1024, 256, 0, stream>>>(Wv, WT + 2 * WSZ);
    wt_kernel<<<1024, 256, 0, stream>>>(Wo, WT + 3 * WSZ);
    ln_kernel<<<32, 128, 0, stream>>>(x, gamma, beta, XN);
    qkv_kernel<<<dim3(64, 8, 3), 128, 0, stream>>>(XN, WT, Qb, Kb, Vt);
    attn_kernel<<<dim3(32, 8, 2), 128, 0, stream>>>(Qb, Kb, Vt, AO);
    oproj_kernel<<<dim3(64, 8), 128, 0, stream>>>(AO, WT + 3 * WSZ, bo, x, out);
}
